// AutoregressiveMultimodalRNN_8607114461756
// MI455X (gfx1250) — matrix operand order repaired and checked by reading, not compiled, not run on hardware
//
#include <hip/hip_runtime.h>
#include <hip/hip_bf16.h>

// ---------------------------------------------------------------------------
// CDNA5 (gfx1250) wave32 WMMA implementation.
// Shapes: N_CHUNKS=8, SEQ=32, BATCH=8, IN=256, HID=128, N_LIN=4, D=4096, M=64.
// ---------------------------------------------------------------------------

typedef __attribute__((ext_vector_type(16))) __bf16 v16bf;
typedef __attribute__((ext_vector_type(8)))  float  v8f;
typedef __attribute__((ext_vector_type(4)))  float  v4f;

// Pack two fp32 into one dword of two bf16 (truncation) with a single
// v_perm_b32: select the high halves of {hi,lo} -> {hi.b3,hi.b2,lo.b3,lo.b2}.
__device__ __forceinline__ unsigned pk_bf16(float lo, float hi) {
  return __builtin_amdgcn_perm(__builtin_bit_cast(unsigned, hi),
                               __builtin_bit_cast(unsigned, lo),
                               0x07060302u);
}

__device__ __forceinline__ float sigm(float x) { return 1.f / (1.f + __expf(-x)); }

// ---------------------------------------------------------------------------
// Generic C[M,N] = A[M,K] @ W[N,K]^T + bias1 + bias2 (both operands K-major).
// One wave per 16x16 output tile; bf16 WMMA, f32 accumulate.
// A-fragment layout (16-bit A 16x32): lane<16 -> K octets {0..7,16..23},
// lane>=16 -> {8..15,24..31}; row m = lane&15.
// B-fragment layout (32x16): same K octets as A (base 8*half, +16); n = lane&15;
// dword j holds the same K pair as A's dword j.
// C layout: v8f element r -> row r + 8*(lane>=16), col = lane&15.
// ---------------------------------------------------------------------------
__global__ __launch_bounds__(128) void gemm_nt_wmma(
    const float* __restrict__ A, const float* __restrict__ W,
    const float* __restrict__ bias1, const float* __restrict__ bias2,
    float* __restrict__ C, int M, int N, int K)
{
  const int lane   = threadIdx.x & 31;
  const int wid    = (blockIdx.x * blockDim.x + threadIdx.x) >> 5;
  const int tilesN = N >> 4;
  const int mtile  = wid / tilesN;
  const int ntile  = wid % tilesN;
  if (mtile >= (M >> 4)) return;

  const int half = lane >> 4;     // 0 or 1
  const int l15  = lane & 15;

  const float* Arow = A + (size_t)(mtile * 16 + l15) * K + half * 8;
  const float* Wrow = W + (size_t)(ntile * 16 + l15) * K + half * 8;

  v8f acc = {0.f, 0.f, 0.f, 0.f, 0.f, 0.f, 0.f, 0.f};

#pragma unroll 2
  for (int k0 = 0; k0 < K; k0 += 32) {
    // Speculative prefetch of the streamed weight operand (branch-free;
    // past-the-end translations are silently dropped on CDNA5).
    __builtin_prefetch(Wrow + k0 + 512, 0, 1);

    v4f a0 = *(const v4f*)(Arow + k0);
    v4f a1 = *(const v4f*)(Arow + k0 + 4);
    v4f a2 = *(const v4f*)(Arow + k0 + 16);
    v4f a3 = *(const v4f*)(Arow + k0 + 20);
    v4f b0 = *(const v4f*)(Wrow + k0);
    v4f b1 = *(const v4f*)(Wrow + k0 + 4);
    v4f b2 = *(const v4f*)(Wrow + k0 + 16);
    v4f b3 = *(const v4f*)(Wrow + k0 + 20);

    union { unsigned u[8]; v16bf v; } au, bu;
    au.u[0] = pk_bf16(a0[0], a0[1]); au.u[1] = pk_bf16(a0[2], a0[3]);
    au.u[2] = pk_bf16(a1[0], a1[1]); au.u[3] = pk_bf16(a1[2], a1[3]);
    au.u[4] = pk_bf16(a2[0], a2[1]); au.u[5] = pk_bf16(a2[2], a2[3]);
    au.u[6] = pk_bf16(a3[0], a3[1]); au.u[7] = pk_bf16(a3[2], a3[3]);
    bu.u[0] = pk_bf16(b0[0], b0[1]); bu.u[1] = pk_bf16(b0[2], b0[3]);
    bu.u[2] = pk_bf16(b1[0], b1[1]); bu.u[3] = pk_bf16(b1[2], b1[3]);
    bu.u[4] = pk_bf16(b2[0], b2[1]); bu.u[5] = pk_bf16(b2[2], b2[3]);
    bu.u[6] = pk_bf16(b3[0], b3[1]); bu.u[7] = pk_bf16(b3[2], b3[3]);

    acc = __builtin_amdgcn_wmma_f32_16x16x32_bf16(
        false, au.v, false, bu.v, (short)0, acc, false, false);
  }

  const int n = ntile * 16 + l15;
  float badd = 0.f;
  if (bias1) badd += bias1[n];
  if (bias2) badd += bias2[n];
  const int mbase = mtile * 16 + half * 8;
#pragma unroll
  for (int r = 0; r < 8; ++r)
    C[(size_t)(mbase + r) * N + n] = acc[r] + badd;
}

// ---------------------------------------------------------------------------
// Serial LSTM over 256 steps. Single workgroup, 512 threads = 16 waves.
// gx[t][b][g] (t=256, b=8, g=512) precomputed by gemm_nt_wmma.
// Whh [512,128] is held entirely in registers as bf16 B-fragments
// (2 N-tiles x 4 K-steps per wave = 64 VGPRs). Per step the recurrent
// h @ Whh^T is 8 WMMAs per wave on a batch-padded (8->16) A tile.
// Per-thread cell state c lives in registers; h lives in LDS.
// hs output layout: [chunk*8+b][s*128 + h]  (row-major [64][4096]).
// ---------------------------------------------------------------------------
__global__ __launch_bounds__(512) void lstm_kernel(
    const float* __restrict__ gx, const float* __restrict__ Whh,
    float* __restrict__ hs)
{
  __shared__ float h_lds[8 * 128];
  __shared__ float g_lds[8 * 512];

  const int tid  = threadIdx.x;
  const int lane = tid & 31;
  const int wave = tid >> 5;      // 0..15
  const int half = lane >> 4;
  const int l15  = lane & 15;

  // Preload Whh bf16 B-fragments into registers.
  v16bf Bfrag[2][4];
  {
    const int kb = half * 16;
#pragma unroll
    for (int nt = 0; nt < 2; ++nt) {
      const float* wr = Whh + (size_t)((wave * 2 + nt) * 16 + l15) * 128 + kb;
#pragma unroll
      for (int ks = 0; ks < 4; ++ks) {
        const float* p = wr + ks * 32;
        union { unsigned u[8]; v16bf v; } bu;
#pragma unroll
        for (int j = 0; j < 8; ++j) bu.u[j] = pk_bf16(p[2 * j], p[2 * j + 1]);
        Bfrag[nt][ks] = bu.v;
      }
    }
  }
  for (int i = tid; i < 8 * 128; i += 512) h_lds[i] = 0.f;
  __syncthreads();

  float cst0 = 0.f, cst1 = 0.f;
  const int am   = l15;        // A row (batch; >=8 is zero padding)
  const int aoff = half * 16;  // K half select

  for (int t = 0; t < 256; ++t) {
    v8f acc0 = {0.f,0.f,0.f,0.f,0.f,0.f,0.f,0.f};
    v8f acc1 = {0.f,0.f,0.f,0.f,0.f,0.f,0.f,0.f};
#pragma unroll
    for (int ks = 0; ks < 4; ++ks) {
      union { unsigned u[8]; v16bf v; } au;
      if (am < 8) {
        const float* hr = h_lds + am * 128 + ks * 32 + aoff;
#pragma unroll
        for (int j = 0; j < 4; ++j) au.u[j]     = pk_bf16(hr[2*j],      hr[2*j + 1]);
#pragma unroll
        for (int j = 0; j < 4; ++j) au.u[4 + j] = pk_bf16(hr[8 + 2*j], hr[8 + 2*j + 1]);
      } else {
#pragma unroll
        for (int j = 0; j < 8; ++j) au.u[j] = 0u;
      }
      acc0 = __builtin_amdgcn_wmma_f32_16x16x32_bf16(
          false, au.v, false, Bfrag[0][ks], (short)0, acc0, false, false);
      acc1 = __builtin_amdgcn_wmma_f32_16x16x32_bf16(
          false, au.v, false, Bfrag[1][ks], (short)0, acc1, false, false);
    }

    // acc rows m=0..7 (the real batch) live in lanes 0..15 (half==0).
    if (half == 0) {
      const float* gxr  = gx + (size_t)t * 8 * 512;
      const int    col0 = wave * 32 + l15;
      const int    col1 = col0 + 16;
#pragma unroll
      for (int b = 0; b < 8; ++b) {
        g_lds[b * 512 + col0] = acc0[b] + gxr[b * 512 + col0];
        g_lds[b * 512 + col1] = acc1[b] + gxr[b * 512 + col1];
      }
    }
    __syncthreads();

    // Gate nonlinearities: 1024 (b,h) pairs over 512 threads (2 each).
    {
      const int p0 = tid,       b0 = p0 >> 7, h0 = p0 & 127;
      const int p1 = tid + 512, b1 = p1 >> 7, h1 = p1 & 127;
      float gi = g_lds[b0 * 512 + h0],       gf = g_lds[b0 * 512 + 128 + h0];
      float gg = g_lds[b0 * 512 + 256 + h0], go = g_lds[b0 * 512 + 384 + h0];
      cst0 = sigm(gf) * cst0 + sigm(gi) * tanhf(gg);
      float hv0 = sigm(go) * tanhf(cst0);
      gi = g_lds[b1 * 512 + h1];       gf = g_lds[b1 * 512 + 128 + h1];
      gg = g_lds[b1 * 512 + 256 + h1]; go = g_lds[b1 * 512 + 384 + h1];
      cst1 = sigm(gf) * cst1 + sigm(gi) * tanhf(gg);
      float hv1 = sigm(go) * tanhf(cst1);

      h_lds[b0 * 128 + h0] = hv0;
      h_lds[b1 * 128 + h1] = hv1;
      const int chunk = t >> 5, s = t & 31;
      hs[(size_t)(chunk * 8 + b0) * 4096 + s * 128 + h0] = hv0;
      hs[(size_t)(chunk * 8 + b1) * 4096 + s * 128 + h1] = hv1;
    }
    __syncthreads();
  }
}

// ---------------------------------------------------------------------------
// r = softmax(q*k, axis=-1) * v per row of 4096. One block per row.
// ---------------------------------------------------------------------------
__global__ __launch_bounds__(256) void attn_kernel(
    const float* __restrict__ q, const float* __restrict__ k,
    const float* __restrict__ v, float* __restrict__ r)
{
  __shared__ float red[256];
  const int row = blockIdx.x, tid = threadIdx.x;
  const float* qr = q + (size_t)row * 4096;
  const float* kr = k + (size_t)row * 4096;
  const float* vr = v + (size_t)row * 4096;

  float s[16];
  float mx = -3.4e38f;
#pragma unroll
  for (int j = 0; j < 16; ++j) {
    const int d = tid + j * 256;
    s[j] = qr[d] * kr[d];
    mx = fmaxf(mx, s[j]);
  }
  red[tid] = mx; __syncthreads();
  for (int off = 128; off > 0; off >>= 1) {
    if (tid < off) red[tid] = fmaxf(red[tid], red[tid + off]);
    __syncthreads();
  }
  mx = red[0]; __syncthreads();

  float sum = 0.f;
#pragma unroll
  for (int j = 0; j < 16; ++j) { s[j] = __expf(s[j] - mx); sum += s[j]; }
  red[tid] = sum; __syncthreads();
  for (int off = 128; off > 0; off >>= 1) {
    if (tid < off) red[tid] += red[tid + off];
    __syncthreads();
  }
  const float inv = 1.f / red[0];
#pragma unroll
  for (int j = 0; j < 16; ++j) {
    const int d = tid + j * 256;
    r[(size_t)row * 4096 + d] = s[j] * inv * vr[d];
  }
}

// ---------------------------------------------------------------------------
// out[row] = sigmoid(dot(r[row], Wout) + bout). One block per row (64 rows).
// ---------------------------------------------------------------------------
__global__ __launch_bounds__(256) void out_kernel(
    const float* __restrict__ r, const float* __restrict__ Wout,
    const float* __restrict__ bout, float* __restrict__ out)
{
  __shared__ float red[256];
  const int row = blockIdx.x, tid = threadIdx.x;
  const float* rr = r + (size_t)row * 4096;
  float sum = 0.f;
  for (int d = tid; d < 4096; d += 256) sum += rr[d] * Wout[d];
  red[tid] = sum; __syncthreads();
  for (int off = 128; off > 0; off >>= 1) {
    if (tid < off) red[tid] += red[tid + off];
    __syncthreads();
  }
  if (tid == 0) out[row] = 1.f / (1.f + __expf(-(red[0] + bout[0])));
}

// ---------------------------------------------------------------------------
// Host launcher.
// ---------------------------------------------------------------------------
extern "C" void kernel_launch(void* const* d_in, const int* in_sizes, int n_in,
                              void* d_out, int out_size, void* d_ws, size_t ws_size,
                              hipStream_t stream) {
  const float* x    = (const float*)d_in[0];   // [8,32,8,256]
  const float* Wih  = (const float*)d_in[1];   // [512,256]
  const float* Whh  = (const float*)d_in[2];   // [512,128]
  const float* bih  = (const float*)d_in[3];   // [512]
  const float* bhh  = (const float*)d_in[4];   // [512]
  const float* Wq   = (const float*)d_in[5];   // [4096,4096]
  const float* bq   = (const float*)d_in[6];
  const float* Wk   = (const float*)d_in[7];
  const float* bk   = (const float*)d_in[8];
  const float* Wv   = (const float*)d_in[9];
  const float* bv   = (const float*)d_in[10];
  const float* Wl   = (const float*)d_in[11];  // [4,4096,4096]
  const float* bl   = (const float*)d_in[12];  // [4,4096]
  const float* Wout = (const float*)d_in[13];  // [1,4096]
  const float* bout = (const float*)d_in[14];  // [1]
  float* out = (float*)d_out;                  // [64]

  char* ws = (char*)d_ws;
  float* gx = (float*)(ws + ((size_t)0 << 20));  // [2048,512]   4 MB
  float* hs = (float*)(ws + ((size_t)4 << 20));  // [64,4096]    1 MB
  float* q  = (float*)(ws + ((size_t)5 << 20));  // [64,4096]
  float* kk = (float*)(ws + ((size_t)6 << 20));  // [64,4096]
  float* vv = (float*)(ws + ((size_t)7 << 20));  // [64,4096]
  float* r0 = (float*)(ws + ((size_t)8 << 20));  // [64,4096]
  float* r1 = (float*)(ws + ((size_t)9 << 20));  // [64,4096]

  // 1) Input projection for all timesteps: gx = x_flat @ Wih^T + (bih+bhh).
  //    M=2048, N=512, K=256 -> 4096 waves -> 1024 blocks of 128 threads.
  gemm_nt_wmma<<<1024, 128, 0, stream>>>(x, Wih, bih, bhh, gx, 2048, 512, 256);

  // 2) Serial LSTM (single workgroup; WMMA recurrent matmul).
  lstm_kernel<<<1, 512, 0, stream>>>(gx, Whh, hs);

  // 3) Q/K/V projections: M=64, N=4096, K=4096 -> 1024 waves -> 256 blocks.
  gemm_nt_wmma<<<256, 128, 0, stream>>>(hs, Wq, bq, nullptr, q,  64, 4096, 4096);
  gemm_nt_wmma<<<256, 128, 0, stream>>>(hs, Wk, bk, nullptr, kk, 64, 4096, 4096);
  gemm_nt_wmma<<<256, 128, 0, stream>>>(hs, Wv, bv, nullptr, vv, 64, 4096, 4096);

  // 4) Elementwise softmax attention.
  attn_kernel<<<64, 256, 0, stream>>>(q, kk, vv, r0);

  // 5) Four chained linear layers (ping-pong r0 <-> r1).
  float* rin = r0; float* rout = r1;
  for (int i = 0; i < 4; ++i) {
    gemm_nt_wmma<<<256, 128, 0, stream>>>(rin, Wl + (size_t)i * 4096 * 4096,
                                          bl + (size_t)i * 4096, nullptr, rout,
                                          64, 4096, 4096);
    float* t = rin; rin = rout; rout = t;
  }

  // 6) Output head: sigmoid(r @ Wout^T + bout).  (rin == r0 after 4 swaps)
  out_kernel<<<64, 256, 0, stream>>>(rin, Wout, bout, out);
}
